// GroupedQueryAttention_71201967833259
// MI455X (gfx1250) — matrix operand order repaired and checked by reading, not compiled, not run on hardware
//
#include <hip/hip_runtime.h>
#include <hip/hip_bf16.h>

#define D_IN 4096
#define NUM_HEADS 32
#define NUM_KV 8
#define HEAD_DIM 128
#define SEQ 2048
#define KV_DIM (NUM_KV * HEAD_DIM)            // 1024
#define NQKV (D_IN + 2 * KV_DIM)              // 6144
#define QSCALE 0.08838834764831845f           // 1/sqrt(128)
#define LN_EPS 1e-6f

typedef __bf16 bf16;
typedef __attribute__((ext_vector_type(16))) __bf16 v16bf;
typedef __attribute__((ext_vector_type(8)))  __bf16 v8bf;
typedef __attribute__((ext_vector_type(8)))  float  v8f;

// ---------------------------------------------------------------------------
// WMMA fragment loaders (CDNA5 wave32 layouts, 16x16x32 bf16)
// A (16x32): lane L holds row M=L%16; elems 0..7 -> K = hi*8 + i,
//            elems 8..15 -> K = 16 + hi*8 + i   (hi = L>=16)
// B (32x16): lane L holds col N=L%16; same K order as A
// C/D (16x16 f32): vgpr r: row = r + hi*8, col = lane%16
// ---------------------------------------------------------------------------
static __device__ __forceinline__ v16bf load_a_frag(const bf16* __restrict__ A,
                                                    int lda, int mBase, int kBase, int lane) {
  int m  = mBase + (lane & 15);
  int hi = (lane >> 4) & 1;
  const bf16* p = A + (size_t)m * lda + kBase + hi * 8;
  v8bf lo = *(const v8bf*)(p);
  v8bf hh = *(const v8bf*)(p + 16);
  v16bf r;
#pragma unroll
  for (int i = 0; i < 8; ++i) { r[i] = lo[i]; r[8 + i] = hh[i]; }
  return r;
}

// Bt is the B matrix stored transposed: Bt[n][k], row stride ldb.
static __device__ __forceinline__ v16bf load_b_frag(const bf16* __restrict__ Bt,
                                                    int ldb, int nBase, int kBase, int lane) {
  int n  = nBase + (lane & 15);
  int hi = (lane >> 4) & 1;
  const bf16* p = Bt + (size_t)n * ldb + kBase + hi * 8;
  v8bf a = *(const v8bf*)(p);
  v8bf b = *(const v8bf*)(p + 16);
  v16bf r;
#pragma unroll
  for (int i = 0; i < 8; ++i) { r[i] = a[i]; r[8 + i] = b[i]; }
  return r;
}

static __device__ __forceinline__ v8f wmma_bf16(v16bf a, v16bf b, v8f c) {
  return __builtin_amdgcn_wmma_f32_16x16x32_bf16(false, a, false, b, (short)0, c, false, false);
}

// ---------------------------------------------------------------------------
// Conversion / packing kernels (one-shot, weights end up L2-resident bf16)
// ---------------------------------------------------------------------------
__global__ void cvt_x_kernel(const float* __restrict__ in, bf16* __restrict__ out, int n) {
  int i = (blockIdx.x * blockDim.x + threadIdx.x) * 4;
  if (i >= n) return;
#pragma unroll
  for (int j = 0; j < 4; ++j) out[i + j] = (bf16)in[i + j];
}

__global__ void pack_wqkv_kernel(const float* __restrict__ Wq, const float* __restrict__ Wk,
                                 const float* __restrict__ Wv, bf16* __restrict__ o) {
  size_t idx = (size_t)blockIdx.x * blockDim.x + threadIdx.x;
  if (idx >= (size_t)NQKV * D_IN) return;
  int n = (int)(idx / D_IN);
  int k = (int)(idx % D_IN);
  float v;
  if (n < D_IN)                v = Wq[(size_t)k * D_IN  + n];
  else if (n < D_IN + KV_DIM)  v = Wk[(size_t)k * KV_DIM + (n - D_IN)];
  else                         v = Wv[(size_t)k * KV_DIM + (n - D_IN - KV_DIM)];
  o[idx] = (bf16)v;
}

__global__ void pack_wo_kernel(const float* __restrict__ Wo, bf16* __restrict__ o) {
  size_t idx = (size_t)blockIdx.x * blockDim.x + threadIdx.x;
  if (idx >= (size_t)D_IN * D_IN) return;
  int n = (int)(idx / D_IN);
  int k = (int)(idx % D_IN);
  o[idx] = (bf16)Wo[(size_t)k * D_IN + n];
}

// ---------------------------------------------------------------------------
// bf16 WMMA GEMM: out[M,N] = A[M,K] * Bt[N,K]^T.
// One wave -> 32x64 (2 M-tiles x 4 N-tiles, B frags reused across M).
// K loop unrolled x2 with ping-pong fragment sets f0/f1: each half-iteration
// loads one set while the 8 WMMAs consume the other, with no register
// rotation copies (K must be a multiple of 64 -- true here: 4096).
// ---------------------------------------------------------------------------
struct Frags { v16bf a0, a1, b0, b1, b2, b3; };

static __device__ __forceinline__ void load_frags(Frags& f,
    const bf16* __restrict__ A, const bf16* __restrict__ Bt,
    int K, int m0, int n0, int k0, int lane) {
  f.a0 = load_a_frag(A, K, m0,      k0, lane);
  f.a1 = load_a_frag(A, K, m0 + 16, k0, lane);
  f.b0 = load_b_frag(Bt, K, n0,      k0, lane);
  f.b1 = load_b_frag(Bt, K, n0 + 16, k0, lane);
  f.b2 = load_b_frag(Bt, K, n0 + 32, k0, lane);
  f.b3 = load_b_frag(Bt, K, n0 + 48, k0, lane);
}

static __device__ __forceinline__ void mma8(const Frags& f, v8f acc[8]) {
  acc[0] = wmma_bf16(f.a0, f.b0, acc[0]);
  acc[1] = wmma_bf16(f.a0, f.b1, acc[1]);
  acc[2] = wmma_bf16(f.a0, f.b2, acc[2]);
  acc[3] = wmma_bf16(f.a0, f.b3, acc[3]);
  acc[4] = wmma_bf16(f.a1, f.b0, acc[4]);
  acc[5] = wmma_bf16(f.a1, f.b1, acc[5]);
  acc[6] = wmma_bf16(f.a1, f.b2, acc[6]);
  acc[7] = wmma_bf16(f.a1, f.b3, acc[7]);
}

__global__ void __launch_bounds__(256)
gemm_bf16_kernel(const bf16* __restrict__ A, const bf16* __restrict__ Bt,
                 float* __restrict__ outF, bf16* __restrict__ outB,
                 int M, int N, int K, int writeF32) {
  int lane = threadIdx.x & 31;
  int wave = blockIdx.x * (blockDim.x >> 5) + (threadIdx.x >> 5);
  int nGroups = N >> 6;                 // 64 columns per wave
  int mTiles  = M >> 5;                 // 32 rows per wave
  if (wave >= mTiles * nGroups) return; // wave-uniform; EXEC stays all-ones
  int m0 = (wave / nGroups) * 32;
  int n0 = (wave % nGroups) * 64;

  v8f acc[8];                           // [j]   -> rows m0..+15,  cols n0+16j
#pragma unroll                          // [4+j] -> rows m0+16..+31
  for (int j = 0; j < 8; ++j)
#pragma unroll
    for (int r = 0; r < 8; ++r) acc[j][r] = 0.0f;

  Frags f0, f1;
  load_frags(f0, A, Bt, K, m0, n0, 0, lane);

  int k0 = 0;
  for (; k0 + 64 < K; k0 += 64) {
    load_frags(f1, A, Bt, K, m0, n0, k0 + 32, lane);
    mma8(f0, acc);
    load_frags(f0, A, Bt, K, m0, n0, k0 + 64, lane);
    mma8(f1, acc);
  }
  // epilogue: f0 holds K-64, load K-32 into f1
  load_frags(f1, A, Bt, K, m0, n0, K - 32, lane);
  mma8(f0, acc);
  mma8(f1, acc);

  int col = lane & 15;
  int rowOff = ((lane >> 4) & 1) * 8;
#pragma unroll
  for (int h = 0; h < 2; ++h)
#pragma unroll
    for (int j = 0; j < 4; ++j)
#pragma unroll
      for (int r = 0; r < 8; ++r) {
        size_t o = (size_t)(m0 + 16 * h + rowOff + r) * N + (n0 + 16 * j + col);
        float v = acc[4 * h + j][r];
        if (writeF32) outF[o] = v;
        else          outB[o] = (bf16)v;
      }
}

// ---------------------------------------------------------------------------
// Per-head-row LayerNorm + RoPE.  One wave per 128-wide row.
// Writes head-major Q (pre-scaled by 1/sqrt(d)), head-major K, transposed V.
// ---------------------------------------------------------------------------
__global__ void __launch_bounds__(256)
ln_rope_kernel(const bf16* __restrict__ qkv,
               const float* __restrict__ cosT, const float* __restrict__ sinT,
               const float* __restrict__ qn_w, const float* __restrict__ qn_b,
               const float* __restrict__ kn_w, const float* __restrict__ kn_b,
               bf16* __restrict__ q, bf16* __restrict__ k, bf16* __restrict__ vT) {
  int lane = threadIdx.x & 31;
  int wave = blockIdx.x * (blockDim.x >> 5) + (threadIdx.x >> 5);
  const int rowsPerSeq = NUM_HEADS + 2 * NUM_KV;  // 48
  if (wave >= SEQ * rowsPerSeq) return;
  int s = wave / rowsPerSeq;
  int j = wave % rowsPerSeq;
  int d0 = lane * 4;

  if (j >= NUM_HEADS + NUM_KV) {  // V: pure bf16 copy into transposed layout
    int g = j - NUM_HEADS - NUM_KV;
    const bf16* src = qkv + (size_t)s * NQKV + D_IN + KV_DIM + g * HEAD_DIM;
#pragma unroll
    for (int i = 0; i < 4; ++i)
      vT[(size_t)g * HEAD_DIM * SEQ + (size_t)(d0 + i) * SEQ + s] = src[d0 + i];
    return;
  }

  bool isQ = (j < NUM_HEADS);
  int srcOff = isQ ? (j * HEAD_DIM) : (D_IN + (j - NUM_HEADS) * HEAD_DIM);
  const float* w = isQ ? qn_w : kn_w;
  const float* b = isQ ? qn_b : kn_b;
  const bf16* src = qkv + (size_t)s * NQKV + srcOff;

  float x[4], sum = 0.f, sq = 0.f;
#pragma unroll
  for (int i = 0; i < 4; ++i) { x[i] = (float)src[d0 + i]; sum += x[i]; sq += x[i] * x[i]; }
#pragma unroll
  for (int m = 16; m >= 1; m >>= 1) { sum += __shfl_xor(sum, m, 32); sq += __shfl_xor(sq, m, 32); }
  float mean = sum * (1.f / HEAD_DIM);
  float var  = sq * (1.f / HEAD_DIM) - mean * mean;
  float rstd = rsqrtf(var + LN_EPS);

  float nv[4];
#pragma unroll
  for (int i = 0; i < 4; ++i) nv[i] = (x[i] - mean) * rstd * w[d0 + i] + b[d0 + i];
  // RoPE: partner element d^64 lives in lane^16 (4 elems/lane, 64/4 = 16)
  float ov[4];
#pragma unroll
  for (int i = 0; i < 4; ++i) ov[i] = __shfl_xor(nv[i], 16, 32);

  bf16* dst = isQ ? (q + ((size_t)j * SEQ + s) * HEAD_DIM)
                  : (k + ((size_t)(j - NUM_HEADS) * SEQ + s) * HEAD_DIM);
#pragma unroll
  for (int i = 0; i < 4; ++i) {
    int d = d0 + i;
    float rot = (d < HEAD_DIM / 2) ? -ov[i] : ov[i];
    float o = nv[i] * cosT[(size_t)s * HEAD_DIM + d] + rot * sinT[(size_t)s * HEAD_DIM + d];
    if (isQ) o *= QSCALE;
    dst[d] = (bf16)o;
  }
}

// ---------------------------------------------------------------------------
// Flash-style causal GQA attention.  Block = 4 waves = 4 heads, one 16-row
// q-tile (uniform trip count -> __syncthreads legal).  Per wave:
//   S(16x32) via 8 WMMAs, online softmax (xor-16 shuffles),
//   P -> LDS bf16 -> A-frag, ctx += P*V via 8 WMMAs.
// ---------------------------------------------------------------------------
__global__ void __launch_bounds__(128)
attn_kernel(const bf16* __restrict__ q, const bf16* __restrict__ k,
            const bf16* __restrict__ vT, bf16* __restrict__ ctx) {
  __shared__ __align__(16) bf16 lds[4][16][32];
  int lane = threadIdx.x & 31;
  int w    = threadIdx.x >> 5;
  int qt   = blockIdx.x;
  int head = blockIdx.y * 4 + w;
  int g    = head >> 2;                 // GROUP_SIZE = 4
  int q0   = qt * 16;

  const bf16* qh = q  + (size_t)head * SEQ * HEAD_DIM;
  const bf16* kh = k  + (size_t)g    * SEQ * HEAD_DIM;
  const bf16* vh = vT + (size_t)g    * HEAD_DIM * SEQ;   // vT[d][s]

  v16bf qf[4];
#pragma unroll
  for (int j = 0; j < 4; ++j) qf[j] = load_a_frag(qh, HEAD_DIM, q0, 32 * j, lane);

  v8f O[8];
  float mrow[8], lrow[8], alpha[8];
#pragma unroll
  for (int t = 0; t < 8; ++t) {
#pragma unroll
    for (int r = 0; r < 8; ++r) O[t][r] = 0.f;
    mrow[t] = -1e30f; lrow[t] = 0.f;
  }

  int rowOff = ((lane >> 4) & 1) * 8;
  int cc = lane & 15;
  int nChunks = (q0 + 15) / 32 + 1;

  for (int c = 0; c < nChunks; ++c) {
    int col0 = c * 32;
    v8f S0, S1;
#pragma unroll
    for (int r = 0; r < 8; ++r) { S0[r] = 0.f; S1[r] = 0.f; }
#pragma unroll
    for (int j = 0; j < 4; ++j) {
      v16bf kb0 = load_b_frag(kh, HEAD_DIM, col0,      32 * j, lane);
      v16bf kb1 = load_b_frag(kh, HEAD_DIM, col0 + 16, 32 * j, lane);
      S0 = wmma_bf16(qf[j], kb0, S0);
      S1 = wmma_bf16(qf[j], kb1, S1);
    }
    if (col0 + 31 > q0) {               // causal mask (diagonal chunk)
      int ca = col0 + cc, cb = ca + 16;
#pragma unroll
      for (int r = 0; r < 8; ++r) {
        int row = q0 + rowOff + r;
        if (ca > row) S0[r] = -1e30f;
        if (cb > row) S1[r] = -1e30f;
      }
    }
    // online softmax: each row spans one 16-lane half
#pragma unroll
    for (int r = 0; r < 8; ++r) {
      float t = fmaxf(S0[r], S1[r]);
#pragma unroll
      for (int m = 8; m >= 1; m >>= 1) t = fmaxf(t, __shfl_xor(t, m, 16));
      float mn = fmaxf(mrow[r], t);
      alpha[r] = __expf(mrow[r] - mn);
      mrow[r] = mn;
      S0[r] = __expf(S0[r] - mn);
      S1[r] = __expf(S1[r] - mn);
      float su = S0[r] + S1[r];
#pragma unroll
      for (int m = 8; m >= 1; m >>= 1) su += __shfl_xor(su, m, 16);
      lrow[r] = lrow[r] * alpha[r] + su;
    }
    __syncthreads();                    // WAR vs previous iteration's reads
#pragma unroll
    for (int r = 0; r < 8; ++r) {       // P (C-layout) -> LDS row-major bf16
      lds[w][rowOff + r][cc]      = (bf16)S0[r];
      lds[w][rowOff + r][cc + 16] = (bf16)S1[r];
    }
    __syncthreads();
    int mA = lane & 15, hi = (lane >> 4) & 1;
    v8bf p0 = *(const v8bf*)&lds[w][mA][hi * 8];
    v8bf p1 = *(const v8bf*)&lds[w][mA][16 + hi * 8];
    v16bf pf;
#pragma unroll
    for (int i = 0; i < 8; ++i) { pf[i] = p0[i]; pf[8 + i] = p1[i]; }
#pragma unroll
    for (int t = 0; t < 8; ++t) {
#pragma unroll
      for (int r = 0; r < 8; ++r) O[t][r] *= alpha[r];
      v16bf vb = load_b_frag(vh, SEQ, t * 16, col0, lane);  // Bt[d][s]
      O[t] = wmma_bf16(pf, vb, O[t]);
    }
  }

  float inv[8];
#pragma unroll
  for (int r = 0; r < 8; ++r) inv[r] = 1.f / lrow[r];
#pragma unroll
  for (int t = 0; t < 8; ++t)
#pragma unroll
    for (int r = 0; r < 8; ++r)
      ctx[(size_t)(q0 + rowOff + r) * D_IN + head * HEAD_DIM + t * 16 + cc] =
          (bf16)(O[t][r] * inv[r]);
}

// ---------------------------------------------------------------------------
extern "C" void kernel_launch(void* const* d_in, const int* in_sizes, int n_in,
                              void* d_out, int out_size, void* d_ws, size_t ws_size,
                              hipStream_t stream) {
  const float* x    = (const float*)d_in[0];
  // d_in[1] = mask (causality computed analytically)
  const float* cosT = (const float*)d_in[2];
  const float* sinT = (const float*)d_in[3];
  const float* Wq   = (const float*)d_in[4];
  const float* Wk   = (const float*)d_in[5];
  const float* Wv   = (const float*)d_in[6];
  const float* Wo   = (const float*)d_in[7];
  const float* qn_w = (const float*)d_in[8];
  const float* qn_b = (const float*)d_in[9];
  const float* kn_w = (const float*)d_in[10];
  const float* kn_b = (const float*)d_in[11];
  float* out = (float*)d_out;

  char* p = (char*)d_ws;
  bf16* x_b   = (bf16*)p; p += (size_t)SEQ * D_IN * 2;
  bf16* wqkvT = (bf16*)p; p += (size_t)NQKV * D_IN * 2;
  bf16* woT   = (bf16*)p; p += (size_t)D_IN * D_IN * 2;
  bf16* qkv_b = (bf16*)p; p += (size_t)SEQ * NQKV * 2;
  bf16* q_b   = (bf16*)p; p += (size_t)NUM_HEADS * SEQ * HEAD_DIM * 2;
  bf16* k_b   = (bf16*)p; p += (size_t)NUM_KV * SEQ * HEAD_DIM * 2;
  bf16* vT_b  = (bf16*)p; p += (size_t)NUM_KV * HEAD_DIM * SEQ * 2;
  bf16* ctx_b = (bf16*)p;

  // 1) conversions / packing
  cvt_x_kernel<<<(SEQ * D_IN / 4 + 255) / 256, 256, 0, stream>>>(x, x_b, SEQ * D_IN);
  pack_wqkv_kernel<<<(int)(((size_t)NQKV * D_IN + 255) / 256), 256, 0, stream>>>(Wq, Wk, Wv, wqkvT);
  pack_wo_kernel<<<(int)(((size_t)D_IN * D_IN + 255) / 256), 256, 0, stream>>>(Wo, woT);

  // 2) fused QKV projection: [2048,4096] x [4096,6144] -> bf16
  {
    int waves = (SEQ / 32) * (NQKV / 64);          // 64 * 96 = 6144 waves
    gemm_bf16_kernel<<<waves / 8, 256, 0, stream>>>(x_b, wqkvT, nullptr, qkv_b,
                                                    SEQ, NQKV, D_IN, 0);
  }

  // 3) per-head LayerNorm + RoPE (+1/sqrt(d) into Q), V transposed
  {
    int waves = SEQ * (NUM_HEADS + 2 * NUM_KV);
    ln_rope_kernel<<<waves / 8, 256, 0, stream>>>(qkv_b, cosT, sinT, qn_w, qn_b,
                                                  kn_w, kn_b, q_b, k_b, vT_b);
  }

  // 4) causal GQA flash attention -> ctx bf16 [2048, 4096]
  attn_kernel<<<dim3(SEQ / 16, NUM_HEADS / 4), 128, 0, stream>>>(q_b, k_b, vT_b, ctx_b);

  // 5) output projection: [2048,4096] x [4096,4096] -> fp32 d_out
  {
    int waves = (SEQ / 32) * (D_IN / 64);          // 64 * 64 = 4096 waves
    gemm_bf16_kernel<<<waves / 8, 256, 0, stream>>>(ctx_b, woT, out, nullptr,
                                                    SEQ, D_IN, D_IN, 1);
  }
}
